// TFN_36369783063090
// MI455X (gfx1250) — matrix operand order repaired and checked by reading, not compiled, not run on hardware
//
#include <hip/hip_runtime.h>
#include <hip/hip_bf16.h>
#include <math.h>

// ---------------------------------------------------------------------------
// TFN forward, fused for MI455X (gfx1250, wave32, WMMA).
//   B=32, N=64, c=16, HID=32, L=3, E=64*63=4032 edges per batch.
// Strategy: fuse the radial MLP GEMM (per-edge 16x1024x32, f16 WMMA, f32 acc)
// with the per-edge W-matrix application so the 528MB/layer "k" tensor never
// touches HBM; aggregate messages with f32 atomics into L2-resident buffers.
// ---------------------------------------------------------------------------

typedef __attribute__((ext_vector_type(16))) _Float16 v16h;
typedef __attribute__((ext_vector_type(8)))  float    v8f;

#define BATCH 32
#define NNODE 64
#define NF    16
#define HID   32
#define NEDGE 4032            // 64*63
#define NTILE 252             // NEDGE/16
#define KCOLS 1024            // 4*NF*NF
#define KSTRIDE 1032          // padded LDS row stride (halfs)
#define EPS   1e-8f

// ---------------- repack radial_w2 -> f16 B fragments ----------------------
// Layout: [layer][tile t=0..63][lane=0..31][p=0..15]
//   lanes 0-15 : N = t*16 + lane,     K = p
//   lanes 16-31: N = t*16 + lane-16,  K = 16 + p
__global__ void tfn_prep_b(const float* __restrict__ rw2, _Float16* __restrict__ wB) {
    int gid = blockIdx.x * 256 + threadIdx.x;          // 3*32768 total
    int l    = gid >> 15;
    int rem  = gid & 32767;
    int t    = rem >> 9;
    int lane = (rem >> 4) & 31;
    int p    = rem & 15;
    int col  = t * 16 + (lane & 15);
    int k    = (lane < 16) ? p : (16 + p);
    wB[gid] = (_Float16)rw2[l * (HID * KCOLS) + k * KCOLS + col];
}

// ---------------- init node features ---------------------------------------
__global__ void tfn_init(const float* __restrict__ z, const float* __restrict__ w_s_in,
                         const float* __restrict__ w_v_in, const float* __restrict__ v,
                         float* __restrict__ s, float* __restrict__ vec) {
    int gid  = blockIdx.x * 256 + threadIdx.x;         // B*N*NF = 32768
    int node = gid >> 4;
    int cc   = gid & 15;
    s[gid] = z[node] * w_s_in[cc];
    float wv = w_v_in[cc];
    #pragma unroll
    for (int d = 0; d < 3; ++d)
        vec[gid * 3 + d] = v[node * 3 + d] * wv;
}

// ---------------- fused edge kernel: radial GEMM (WMMA) + W-apply ----------
__global__ void __launch_bounds__(256)
tfn_edge(const float* __restrict__ pos,
         const float* __restrict__ s, const float* __restrict__ vec,
         const _Float16* __restrict__ wBl,
         const float* __restrict__ w1l, const float* __restrict__ b1l,
         const float* __restrict__ b2l,
         float* __restrict__ agg_s, float* __restrict__ agg_v) {
    __shared__ _Float16 kbuf[16 * KSTRIDE];   // k for 16 edges (padded)
    __shared__ float    feat_s[16 * NF];      // s_src staged
    __shared__ float    feat_v[16 * NF * 3];  // v_src staged
    __shared__ float    u_s[16 * 3];
    __shared__ float    r_s[16];
    __shared__ int      src_s[16], dst_s[16];

    const int tid  = threadIdx.x;
    const int b    = blockIdx.y;
    const int e0   = blockIdx.x * 16;

    // Phase 0: edge geometry (16 threads)
    if (tid < 16) {
        int e    = e0 + tid;
        int srci = e / 63;
        int dr   = e % 63;
        int dsti = dr + (dr >= srci ? 1 : 0);
        const float* ps = pos + (b * NNODE + srci) * 3;
        const float* pd = pos + (b * NNODE + dsti) * 3;
        float dx = pd[0] - ps[0], dy = pd[1] - ps[1], dz = pd[2] - ps[2];
        float rr = sqrtf(dx * dx + dy * dy + dz * dz);
        float iv = 1.0f / (rr + EPS);
        u_s[tid * 3 + 0] = dx * iv;
        u_s[tid * 3 + 1] = dy * iv;
        u_s[tid * 3 + 2] = dz * iv;
        r_s[tid]  = rr;
        src_s[tid] = srci;
        dst_s[tid] = dsti;
    }
    __syncthreads();

    // Phase 1a: stage source features (256 threads; src mostly uniform/tile)
    {
        int eloc = tid >> 4, cc = tid & 15;
        int src  = src_s[eloc];
        feat_s[eloc * NF + cc] = s[(b * NNODE + src) * NF + cc];
        #pragma unroll
        for (int d = 0; d < 3; ++d)
            feat_v[(eloc * NF + cc) * 3 + d] = vec[((b * NNODE + src) * NF + cc) * 3 + d];
    }

    // Phase 1b: per-lane A fragment: h = relu(r*w1 + b1), f16 (16x32 A tile)
    const int lane = tid & 31;
    const int row  = lane & 15;
    const float rr = r_s[row];
    const int kb   = (lane < 16) ? 0 : 16;
    v16h afrag;
    #pragma unroll
    for (int q = 0; q < 8; ++q) {
        float h0 = fmaxf(rr * w1l[kb + q]      + b1l[kb + q],      0.0f);
        float h1 = fmaxf(rr * w1l[kb + 8 + q] + b1l[kb + 8 + q], 0.0f);
        afrag[q]     = (_Float16)h0;
        afrag[q + 8] = (_Float16)h1;
    }

    // Phase 2: radial GEMM (16 x 1024 x 32) via 8 WMMAs per wave, k -> LDS
    const int wv = tid >> 5;                  // 8 waves, 8 N-tiles each
    const int mb = (lane < 16) ? 0 : 8;
    #pragma unroll
    for (int q = 0; q < 8; ++q) {
        int t = wv * 8 + q;
        const v16h bfrag = *(const v16h*)(wBl + (size_t)(t * 32 + lane) * 16);
        v8f acc = {};
        acc = __builtin_amdgcn_wmma_f32_16x16x32_f16(
            /*neg_a=*/false, afrag, /*neg_b=*/false, bfrag,
            /*c_mod=*/(short)0, acc, /*reuse_a=*/false, /*reuse_b=*/false);
        int col  = t * 16 + row;
        float bv = b2l[col];
        #pragma unroll
        for (int g = 0; g < 8; ++g)
            kbuf[(mb + g) * KSTRIDE + col] = (_Float16)(acc[g] + bv);
    }
    __syncthreads();

    // Phase 3: apply Wss/Wsv/Wvs/Wvv, scatter-add to dst (16 thr per edge)
    {
        int eloc = tid >> 4, i = tid & 15;
        const _Float16* krow = kbuf + eloc * KSTRIDE;
        const float ux = u_s[eloc * 3 + 0], uy = u_s[eloc * 3 + 1], uz = u_s[eloc * 3 + 2];
        float ms = 0.f, msv = 0.f, mv0 = 0.f, mv1 = 0.f, mv2 = 0.f;
        #pragma unroll
        for (int j = 0; j < NF; ++j) {
            float sj  = feat_s[eloc * NF + j];
            float vj0 = feat_v[(eloc * NF + j) * 3 + 0];
            float vj1 = feat_v[(eloc * NF + j) * 3 + 1];
            float vj2 = feat_v[(eloc * NF + j) * 3 + 2];
            float vdot = vj0 * ux + vj1 * uy + vj2 * uz;
            float wss = (float)krow[0   + i * 16 + j];
            float wsv = (float)krow[256 + i * 16 + j];
            float wvs = (float)krow[512 + i * 16 + j];
            float wvv = (float)krow[768 + i * 16 + j];
            ms  += wss * sj + wvs * vdot;
            msv += wsv * sj;
            mv0 += wvv * vj0;
            mv1 += wvv * vj1;
            mv2 += wvv * vj2;
        }
        mv0 += msv * ux; mv1 += msv * uy; mv2 += msv * uz;
        int dst = dst_s[eloc];
        float* as = agg_s + (b * NNODE + dst) * NF + i;
        float* av = agg_v + ((b * NNODE + dst) * NF + i) * 3;
        unsafeAtomicAdd(as,     ms);
        unsafeAtomicAdd(av + 0, mv0);
        unsafeAtomicAdd(av + 1, mv1);
        unsafeAtomicAdd(av + 2, mv2);
    }
}

// ---------------- node self-interaction + residual --------------------------
__global__ void tfn_node(const float* __restrict__ agg_s, const float* __restrict__ agg_v,
                         const float* __restrict__ wsl, const float* __restrict__ wvl,
                         float* __restrict__ s, float* __restrict__ vec) {
    int gid  = blockIdx.x * 256 + threadIdx.x;   // B*N*NF = 32768
    int base = gid >> 4;
    int i    = gid & 15;
    const float* as = agg_s + base * NF;
    float acc = 0.f;
    #pragma unroll
    for (int c = 0; c < NF; ++c)
        acc += as[c] * wsl[c * NF + i];
    s[gid] = fmaxf(acc, 0.f) + s[gid];
    float a0 = 0.f, a1 = 0.f, a2 = 0.f;
    #pragma unroll
    for (int c = 0; c < NF; ++c) {
        float w = wvl[c * NF + i];
        const float* av = agg_v + (base * NF + c) * 3;
        a0 += av[0] * w; a1 += av[1] * w; a2 += av[2] * w;
    }
    vec[gid * 3 + 0] += a0;
    vec[gid * 3 + 1] += a1;
    vec[gid * 3 + 2] += a2;
}

// ---------------- output: contract vec channels + residual pos --------------
__global__ void tfn_out(const float* __restrict__ vec, const float* __restrict__ wvo,
                        const float* __restrict__ pos, float* __restrict__ out) {
    int gid  = blockIdx.x * 256 + threadIdx.x;   // B*N*3 = 6144
    int node = gid / 3;
    int d    = gid % 3;
    float acc = pos[gid];
    #pragma unroll
    for (int c = 0; c < NF; ++c)
        acc += vec[(node * NF + c) * 3 + d] * wvo[c];
    out[gid] = acc;
}

// ---------------------------------------------------------------------------
extern "C" void kernel_launch(void* const* d_in, const int* in_sizes, int n_in,
                              void* d_out, int out_size, void* d_ws, size_t ws_size,
                              hipStream_t stream) {
    const float* pos    = (const float*)d_in[0];
    const float* v      = (const float*)d_in[1];
    const float* z      = (const float*)d_in[2];
    const float* w_s_in = (const float*)d_in[3];
    const float* w_v_in = (const float*)d_in[4];
    const float* rw1    = (const float*)d_in[5];   // (3,1,32)
    const float* rb1    = (const float*)d_in[6];   // (3,32)
    const float* rw2    = (const float*)d_in[7];   // (3,32,1024)
    const float* rb2    = (const float*)d_in[8];   // (3,1024)
    const float* sws    = (const float*)d_in[9];   // (3,16,16)
    const float* swv    = (const float*)d_in[10];  // (3,16,16)
    const float* wvo    = (const float*)d_in[11];  // (16,1)
    float* out          = (float*)d_out;

    // workspace layout (floats)
    float* s     = (float*)d_ws;                 // 32768
    float* vec   = s + 32768;                    // 98304
    float* agg_s = vec + 98304;                  // 32768
    float* agg_v = agg_s + 32768;                // 98304
    _Float16* wB = (_Float16*)(agg_v + 98304);   // 3*32768 halfs

    tfn_prep_b<<<384, 256, 0, stream>>>(rw2, wB);
    tfn_init<<<128, 256, 0, stream>>>(z, w_s_in, w_v_in, v, s, vec);

    for (int l = 0; l < 3; ++l) {
        hipMemsetAsync(agg_s, 0, (size_t)(32768 + 98304) * sizeof(float), stream);
        tfn_edge<<<dim3(NTILE, BATCH), 256, 0, stream>>>(
            pos, s, vec,
            wB + (size_t)l * 32768,
            rw1 + l * HID, rb1 + l * HID, rb2 + l * KCOLS,
            agg_s, agg_v);
        tfn_node<<<128, 256, 0, stream>>>(agg_s, agg_v, sws + l * 256, swv + l * 256, s, vec);
    }

    tfn_out<<<24, 256, 0, stream>>>(vec, wvo, pos, out);
}
